// PCLADEncoder_43447889166697
// MI455X (gfx1250) — matrix operand order repaired and checked by reading, not compiled, not run on hardware
//
#include <hip/hip_runtime.h>
#include <hip/hip_bf16.h>
#include <stdint.h>

typedef __bf16 bf16;
typedef __attribute__((ext_vector_type(16))) __bf16 v16bf;
typedef __attribute__((ext_vector_type(8)))  float  v8f;
typedef int v4i __attribute__((ext_vector_type(4)));

#define BM 128
#define BN 128
#define BKK 32
#define LDSS 40   // BK + 8 pad (elements); 80B row stride -> conflict-free b128 frag reads
#define BUFSZ (2 * BM * LDSS)   // one double-buffer slot: A tile + B tile (elements)

// ---- CDNA5 async global->LDS path (guarded; falls back to sync copy) ----
#if defined(__has_builtin)
# if __has_builtin(__builtin_amdgcn_global_load_async_to_lds_b128) && \
     __has_builtin(__builtin_amdgcn_s_wait_asynccnt)
#  define USE_ASYNC_LDS 1
# endif
#endif
#ifndef USE_ASYNC_LDS
# define USE_ASYNC_LDS 0
#endif

union FragU { v16bf v; uint4 q[2]; };

// C[M x N] = op(A[M x K] @ B[K x N]) with B supplied TRANSPOSED: Bt is [N][K] row-major.
// A bf16 [M][K]; C row stride = ldc.
template<bool RELU, typename OutT>
__global__ __launch_bounds__(256) void gemm_bf16_wmma(
    const bf16* __restrict__ A,
    const bf16* __restrict__ Bt,
    OutT* __restrict__ C,
    int M, int N, int K, int ldc)
{
    __shared__ bf16 smem[2 * BUFSZ];   // [buf][A(128x40) | B(128x40)]

    const int tid  = threadIdx.x;
    const int lane = tid & 31;
    const int wid  = tid >> 5;      // 0..7
    const int wm   = wid >> 1;      // 0..3  -> 32-row strip
    const int wn   = wid & 1;       // 0..1  -> 64-col strip
    const int half = lane >> 4;     // 0/1
    const int l15  = lane & 15;

    const int blockM = blockIdx.y * BM;
    const int blockN = blockIdx.x * BN;

    // staging: each thread owns one 32-byte chunk of the A tile and one of the B tile
    const int s_row = tid >> 1;          // 0..127
    const int s_kg  = (tid & 1) * 16;    // 0 / 16

    const bf16* gA = A  + (size_t)(blockM + s_row) * K + s_kg;   // + k0 per stage
    const bf16* gB = Bt + (size_t)(blockN + s_row) * K + s_kg;
    const int sOff = s_row * LDSS + s_kg;

    v8f acc[2][4];
#pragma unroll
    for (int mi = 0; mi < 2; ++mi)
#pragma unroll
        for (int ni = 0; ni < 4; ++ni)
            acc[mi][ni] = (v8f){0.f,0.f,0.f,0.f,0.f,0.f,0.f,0.f};

    auto stage = [&](int buf, int k0) {
        bf16* dA = &smem[buf * BUFSZ + sOff];
        bf16* dB = dA + BM * LDSS;
        const bf16* sa = gA + k0;
        const bf16* sb = gB + k0;
#if USE_ASYNC_LDS
        __builtin_amdgcn_global_load_async_to_lds_b128(
            (__attribute__((address_space(1))) v4i*)sa,
            (__attribute__((address_space(3))) v4i*)dA, 0, 0);
        __builtin_amdgcn_global_load_async_to_lds_b128(
            (__attribute__((address_space(1))) v4i*)sa,
            (__attribute__((address_space(3))) v4i*)dA, 16, 0);
        __builtin_amdgcn_global_load_async_to_lds_b128(
            (__attribute__((address_space(1))) v4i*)sb,
            (__attribute__((address_space(3))) v4i*)dB, 0, 0);
        __builtin_amdgcn_global_load_async_to_lds_b128(
            (__attribute__((address_space(1))) v4i*)sb,
            (__attribute__((address_space(3))) v4i*)dB, 16, 0);
#else
        uint4 a0 = ((const uint4*)sa)[0];
        uint4 a1 = ((const uint4*)sa)[1];
        uint4 b0 = ((const uint4*)sb)[0];
        uint4 b1 = ((const uint4*)sb)[1];
        *(uint4*)(dA)     = a0;
        *(uint4*)(dA + 8) = a1;
        *(uint4*)(dB)     = b0;
        *(uint4*)(dB + 8) = b1;
#endif
    };

    stage(0, 0);
    int cur = 0;

    for (int k0 = 0; k0 < K; k0 += BKK) {
        if (k0 + BKK < K) {
            stage(cur ^ 1, k0 + BKK);       // prefetch next K-step into other buffer
#if USE_ASYNC_LDS
            __builtin_amdgcn_s_wait_asynccnt(4);   // newest 4 may fly; current stage done
#endif
        } else {
#if USE_ASYNC_LDS
            __builtin_amdgcn_s_wait_asynccnt(0);
#endif
        }
        __syncthreads();

        const bf16* ldsA = &smem[cur * BUFSZ];
        const bf16* ldsB = ldsA + BM * LDSS;

        // ---- fragment loads (contiguous 16B chunks matching ISA VGPR layout)
        FragU afr[2], bfr[4];
#pragma unroll
        for (int mi = 0; mi < 2; ++mi) {
            const int r = wm * 32 + mi * 16 + l15;
            afr[mi].q[0] = *(const uint4*)(&ldsA[r * LDSS + half * 8]);        // K = half*8 + 0..7
            afr[mi].q[1] = *(const uint4*)(&ldsA[r * LDSS + 16 + half * 8]);   // K = 16 + half*8 + 0..7
        }
#pragma unroll
        for (int ni = 0; ni < 4; ++ni) {
            const int c = wn * 64 + ni * 16 + l15;
            bfr[ni].q[0] = *(const uint4*)(&ldsB[c * LDSS + half * 8]);        // K = half*8 + 0..7
            bfr[ni].q[1] = *(const uint4*)(&ldsB[c * LDSS + 16 + half * 8]);   // K = 16 + half*8 + 0..7
        }

        // ---- 8x v_wmma_f32_16x16x32_bf16
#pragma unroll
        for (int mi = 0; mi < 2; ++mi)
#pragma unroll
            for (int ni = 0; ni < 4; ++ni)
                acc[mi][ni] = __builtin_amdgcn_wmma_f32_16x16x32_bf16(
                    false, afr[mi].v, false, bfr[ni].v,
                    (short)0, acc[mi][ni], false, false);

        __syncthreads();   // protect WAR: buffer `cur` is restaged next iteration
        cur ^= 1;
    }

    // ---- epilogue: C/D layout -> VGPR r holds M = r + 8*half, N = l15 (u32 address math)
    const uint32_t cbase = (uint32_t)(blockM + wm * 32 + half * 8) * (uint32_t)ldc
                         + (uint32_t)(blockN + wn * 64 + l15);
    OutT* Cp = C + cbase;
#pragma unroll
    for (int mi = 0; mi < 2; ++mi) {
#pragma unroll
        for (int ni = 0; ni < 4; ++ni) {
#pragma unroll
            for (int r = 0; r < 8; ++r) {
                float v = acc[mi][ni][r];
                if (RELU) v = fmaxf(v, 0.0f);
                Cp[(uint32_t)(mi * 16 + r) * (uint32_t)ldc + (uint32_t)(ni * 16)] = (OutT)v;
            }
        }
    }
}

__global__ void f32_to_bf16_kern(const float* __restrict__ in, bf16* __restrict__ out, long long n) {
    long long i = (long long)blockIdx.x * blockDim.x + threadIdx.x;
    if (i < n) out[i] = (bf16)in[i];
}

// Batched convert+transpose: in[b] is [K][N] f32, out[b] is [N][K] bf16. grid.y = batch.
__global__ void f32_to_bf16_T_kern(const float* __restrict__ in, bf16* __restrict__ out, int K, int N) {
    const float* src = in  + (size_t)blockIdx.y * K * N;
    bf16*       dst  = out + (size_t)blockIdx.y * K * N;
    long long i = (long long)blockIdx.x * blockDim.x + threadIdx.x;
    if (i < (long long)K * N) {
        int k = (int)(i / N);
        int n = (int)(i % N);
        dst[(size_t)n * K + k] = (bf16)src[i];
    }
}

// x_cat[b][0][f] = mean_t x_cat[b][1+t][f]
__global__ void mean_slot0_kern(bf16* __restrict__ xcat, int Bn, int F, int T) {
    long long i = (long long)blockIdx.x * blockDim.x + threadIdx.x;
    long long n = (long long)Bn * F;
    if (i >= n) return;
    int b = (int)(i / F);
    int f = (int)(i % F);
    size_t base = (size_t)b * (T + 1) * F;
    float s = 0.f;
    for (int t = 0; t < T; ++t) s += (float)xcat[base + (size_t)(t + 1) * F + f];
    xcat[base + f] = (bf16)(s / (float)T);
}

extern "C" void kernel_launch(void* const* d_in, const int* in_sizes, int n_in,
                              void* d_out, int out_size, void* d_ws, size_t ws_size,
                              hipStream_t stream) {
    (void)in_sizes; (void)n_in; (void)out_size; (void)ws_size;

    constexpr int Bn = 8192, F = 512, T = 11, H = 1024, EH = 1024, Z = 256;
    const int M12 = Bn * (T + 1);   // 98304

    const float* x   = (const float*)d_in[0];
    const float* tW1 = (const float*)d_in[1];
    const float* tW2 = (const float*)d_in[2];
    const float* eW0 = (const float*)d_in[3];
    const float* eW1 = (const float*)d_in[4];
    const float* eW2 = (const float*)d_in[5];
    const float* eW3 = (const float*)d_in[6];
    const float* eW4 = (const float*)d_in[7];
    float* out = (float*)d_out;

    char* ws = (char*)d_ws;
    size_t off = 0;
    auto alloc = [&](size_t bytes) -> char* {
        char* p = ws + off;
        off += (bytes + 255) & ~(size_t)255;
        return p;
    };

    bf16* xb   = (bf16*)alloc((size_t)Bn * F * 2);
    bf16* tW1t = (bf16*)alloc((size_t)T * F * H * 2);   // per t: [H][F]
    bf16* tW2t = (bf16*)alloc((size_t)T * H * F * 2);   // per t: [F][H]
    bf16* eW0t = (bf16*)alloc((size_t)F * EH * 2);      // [EH][F]
    bf16* eW1t = (bf16*)alloc((size_t)EH * EH * 2);
    bf16* eW2t = (bf16*)alloc((size_t)EH * EH * 2);
    bf16* eW3t = (bf16*)alloc((size_t)EH * EH * 2);
    bf16* eW4t = (bf16*)alloc((size_t)EH * Z * 2);      // [Z][EH]
    bf16* hbuf = (bf16*)alloc((size_t)Bn * H * 2);
    bf16* xcat = (bf16*)alloc((size_t)M12 * F * 2);
    bf16* zA   = (bf16*)alloc((size_t)M12 * EH * 2);
    bf16* zB   = (bf16*)alloc((size_t)M12 * EH * 2);

    // activations: plain convert
    {
        long long n = (long long)Bn * F;
        f32_to_bf16_kern<<<dim3((unsigned)((n + 255) / 256)), dim3(256), 0, stream>>>(x, xb, n);
    }
    // weights: convert + transpose ([K][N] -> [N][K]) once; GEMM then streams both
    // operands as contiguous rows (async-to-LDS friendly, no in-kernel transpose)
    auto cvtT = [&](const float* src, bf16* dst, int K, int N, int batch) {
        long long n = (long long)K * N;
        f32_to_bf16_T_kern<<<dim3((unsigned)((n + 255) / 256), batch), dim3(256), 0, stream>>>(src, dst, K, N);
    };
    cvtT(tW1, tW1t, F,  H,  T);
    cvtT(tW2, tW2t, H,  F,  T);
    cvtT(eW0, eW0t, F,  EH, 1);
    cvtT(eW1, eW1t, EH, EH, 1);
    cvtT(eW2, eW2t, EH, EH, 1);
    cvtT(eW3, eW3t, EH, EH, 1);
    cvtT(eW4, eW4t, EH, Z,  1);

    // Transforms: h = relu(x @ tW1[t]); x_cat[:, t+1, :] = h @ tW2[t]
    for (int t = 0; t < T; ++t) {
        gemm_bf16_wmma<true, bf16><<<dim3(H / BN, Bn / BM), dim3(256), 0, stream>>>(
            xb, tW1t + (size_t)t * F * H, hbuf, Bn, H, F, H);
        gemm_bf16_wmma<false, bf16><<<dim3(F / BN, Bn / BM), dim3(256), 0, stream>>>(
            hbuf, tW2t + (size_t)t * H * F, xcat + (size_t)(t + 1) * F, Bn, F, H, (T + 1) * F);
    }

    // Slot 0 = mean over T transforms
    {
        long long n = (long long)Bn * F;
        mean_slot0_kern<<<dim3((unsigned)((n + 255) / 256)), dim3(256), 0, stream>>>(xcat, Bn, F, T);
    }

    // Encoder: 4 hidden ReLU layers + linear output
    gemm_bf16_wmma<true, bf16><<<dim3(EH / BN, M12 / BM), dim3(256), 0, stream>>>(
        xcat, eW0t, zA, M12, EH, F, EH);
    gemm_bf16_wmma<true, bf16><<<dim3(EH / BN, M12 / BM), dim3(256), 0, stream>>>(
        zA, eW1t, zB, M12, EH, EH, EH);
    gemm_bf16_wmma<true, bf16><<<dim3(EH / BN, M12 / BM), dim3(256), 0, stream>>>(
        zB, eW2t, zA, M12, EH, EH, EH);
    gemm_bf16_wmma<true, bf16><<<dim3(EH / BN, M12 / BM), dim3(256), 0, stream>>>(
        zA, eW3t, zB, M12, EH, EH, EH);
    gemm_bf16_wmma<false, float><<<dim3(Z / BN, M12 / BM), dim3(256), 0, stream>>>(
        zB, eW4t, out, M12, Z, EH, Z);
}
